// C4Transformer_62380105007440
// MI455X (gfx1250) — matrix operand order repaired and checked by reading, not compiled, not run on hardware
//
#include <hip/hip_runtime.h>
#include <hip/hip_bf16.h>

// ---------------------------------------------------------------------------
// MI455X (gfx1250) implementation of the byte-level top-k transformer.
// DV = DIM/HEADS = 1  ->  q,k,v are scalars per (token, head).
// Memory-bound workload (~0.25 TFLOP vs 134MB/tensor activations @ 23.3TB/s),
// so: bf16 WMMA GEMMs (f32 accum), f32 residual stream, fused swiglu epilogue,
// A-fragments held in registers across all column tiles (A read once).
// ---------------------------------------------------------------------------

typedef __bf16 bf16_t;
typedef __attribute__((ext_vector_type(16))) __bf16 v16bf;
typedef __attribute__((ext_vector_type(8)))  __bf16 v8bf;
typedef __attribute__((ext_vector_type(4)))  __bf16 v4bf;
typedef __attribute__((ext_vector_type(8)))  float  v8f;

#define NTOK   262144
#define DIMF   128
#define HEADS  128
#define TOPK   32
#define BLOCKN 256
#define NBLK   1024
#define HIDRAW 341
#define HIDPAD 352          // 341 padded to 22*16
#define EPSF   1.1920929e-07f

// ---------------------------------------------------------------------------
// WMMA fragment loaders (CDNA5 ISA 7.12.2 layouts, bf16 16x16x32).
// A (16x32, MxK), row-major A in memory:
//   lane L<16 : row M=L,     K chunks {kb..kb+7} and {kb+16..kb+23}
//   lane L>=16: row M=L-16,  K chunks {kb+8..kb+15} and {kb+24..kb+31}
// B (32x16, KxN) from row-major W (N x K)  (B = W^T):
//   lane L<16 : col N=L,     K chunks {kb..kb+7} and {kb+16..kb+23}
//   lane L>=16: col N=L-16,  K chunks {kb+8..kb+15} and {kb+24..kb+31}
// C/D f32 16x16: vgpr r, lane L<16 -> (M=r, N=L); lane>=16 -> (M=8+r, N=L-16)
// ---------------------------------------------------------------------------
__device__ __forceinline__ v16bf load_a_frag(const bf16_t* __restrict__ A, int lda,
                                             int mBase, int kBase, int lane) {
  const bf16_t* p = A + (size_t)(mBase + (lane & 15)) * lda
                      + kBase + ((lane & 16) ? 8 : 0);
  union { v16bf v; v8bf h[2]; } u;
  u.h[0] = *(const v8bf*)(p);
  u.h[1] = *(const v8bf*)(p + 16);
  return u.v;
}

__device__ __forceinline__ v16bf load_b_frag(const bf16_t* __restrict__ W, int ldw,
                                             int nBase, int kBase, int lane) {
  const bf16_t* p = W + (size_t)(nBase + (lane & 15)) * ldw
                      + kBase + ((lane & 16) ? 8 : 0);
  union { v16bf v; v8bf h[2]; } u;
  u.h[0] = *(const v8bf*)(p);
  u.h[1] = *(const v8bf*)(p + 16);
  return u.v;
}

__device__ __forceinline__ v8f wmma_bf16(v16bf a, v16bf b, v8f c) {
  return __builtin_amdgcn_wmma_f32_16x16x32_bf16(false, a, false, b,
                                                 (short)0, c, false, false);
}

// ---------------------------------------------------------------------------
// Generic GEMM:  D(n x NDIM) = A(n x KDIM, bf16) * W^T (W is NDIM x KDIM bf16)
// MODE 0: store f32 to out (ldout)
// MODE 1: residual add into out (f32, ld=128)
// MODE 2: gate/up fused swiglu: W holds 704x128 (gate rows 0..351, up 352..703),
//         writes bf16 silu(gate)*up to outbf (ld = NDIM = 352)
// Block = 256 threads = 8 waves; wave w owns 16 rows; A frags live in VGPRs
// across the whole column loop (A read once from HBM).
// ---------------------------------------------------------------------------
template<int KDIM, int NDIM, int MODE>
__global__ void gemm_kernel(const bf16_t* __restrict__ A,
                            const bf16_t* __restrict__ W,
                            float* __restrict__ out,
                            bf16_t* __restrict__ outbf,
                            int ldout) {
  constexpr int KT = KDIM / 32;
  const int lane  = threadIdx.x & 31;
  const int wave  = threadIdx.x >> 5;
  const int mBase = (blockIdx.x * 8 + wave) * 16;

  v16bf afr[KT];
#pragma unroll
  for (int t = 0; t < KT; ++t)
    afr[t] = load_a_frag(A, KDIM, mBase, t * 32, lane);

  const int hi8 = (lane & 16) ? 8 : 0;
  const int col = lane & 15;

  for (int j = 0; j < NDIM / 16; ++j) {
    const int nBase = j * 16;
    v8f acc = {};
#pragma unroll
    for (int t = 0; t < KT; ++t)
      acc = wmma_bf16(afr[t], load_b_frag(W, KDIM, nBase, t * 32, lane), acc);

    if (MODE == 2) {
      v8f acc2 = {};
#pragma unroll
      for (int t = 0; t < KT; ++t)
        acc2 = wmma_bf16(afr[t], load_b_frag(W, KDIM, HIDPAD + nBase, t * 32, lane), acc2);
#pragma unroll
      for (int r = 0; r < 8; ++r) {
        const size_t row = (size_t)(mBase + r + hi8);
        float g = acc[r], u = acc2[r];
        float h = (g / (1.f + __expf(-g))) * u;   // silu(g) * u
        outbf[row * NDIM + nBase + col] = (bf16_t)h;
      }
    } else if (MODE == 1) {
#pragma unroll
      for (int r = 0; r < 8; ++r) {
        const size_t idx = (size_t)(mBase + r + hi8) * 128 + nBase + col;
        out[idx] += acc[r];
      }
    } else {
#pragma unroll
      for (int r = 0; r < 8; ++r) {
        const size_t idx = (size_t)(mBase + r + hi8) * ldout + nBase + col;
        out[idx] = acc[r];
      }
    }
  }
}

// ---------------------------------------------------------------------------
// Embedding: x[n][:] = byte_embed[bytes[n]][:] + block_pos[n>>8][:] + local_pos[n&255][:]
// ---------------------------------------------------------------------------
__global__ void embed_kernel(const int* __restrict__ bytes,
                             const float* __restrict__ be,
                             const float* __restrict__ bpe,
                             const float* __restrict__ lpe,
                             float* __restrict__ x) {
  const size_t tid = (size_t)blockIdx.x * blockDim.x + threadIdx.x;
  const size_t n = tid >> 5;
  const int    c = (int)(tid & 31) * 4;
  if (n >= NTOK) return;
  const int b = bytes[n];
  float4 e = *(const float4*)(be  + (size_t)b * DIMF + c);
  float4 p = *(const float4*)(bpe + (size_t)(n >> 8) * DIMF + c);
  float4 l = *(const float4*)(lpe + (size_t)(n & 255) * DIMF + c);
  float4 r; r.x = e.x + p.x + l.x; r.y = e.y + p.y + l.y;
            r.z = e.z + p.z + l.z; r.w = e.w + p.w + l.w;
  *(float4*)(x + n * DIMF + c) = r;
}

// ---------------------------------------------------------------------------
// RMSNorm row (128 cols): one wave per row, lane holds 4 floats, shfl reduce.
// Output bf16 (GEMM A operand).
// ---------------------------------------------------------------------------
__global__ void rmsnorm_kernel(const float* __restrict__ x,
                               const float* __restrict__ w,
                               bf16_t* __restrict__ xn) {
  const int lane = threadIdx.x & 31;
  const int wave = threadIdx.x >> 5;
  const size_t row = (size_t)blockIdx.x * 8 + wave;
  const float* xr = x + row * DIMF;
  float4 v  = ((const float4*)xr)[lane];
  float ss = v.x * v.x + v.y * v.y + v.z * v.z + v.w * v.w;
#pragma unroll
  for (int off = 16; off > 0; off >>= 1)
    ss += __shfl_xor(ss, off, 32);
  const float scale = rsqrtf(ss * (1.f / 128.f) + EPSF);
  float4 wv = ((const float4*)w)[lane];
  v4bf o;
  o[0] = (bf16_t)(v.x * scale * wv.x);
  o[1] = (bf16_t)(v.y * scale * wv.y);
  o[2] = (bf16_t)(v.z * scale * wv.z);
  o[3] = (bf16_t)(v.w * scale * wv.w);
  *(v4bf*)(xn + row * DIMF + lane * 4) = o;
}

// ---------------------------------------------------------------------------
// Per-head global top-32 / bottom-32 key selection over all 262144 tokens.
// One block (64 threads) per head. Each thread keeps a sorted-32 list in LDS
// (insert guarded by compare vs current 32nd -> rare after warmup), then a
// serial 64-way merge picks the final 32 (2K scalar ops, negligible).
// Values are gathered by index afterwards (v lives at qkv col 256+h).
// ---------------------------------------------------------------------------
__global__ void topk_kernel(const float* __restrict__ qkv,
                            float* __restrict__ topkey, float* __restrict__ topval,
                            float* __restrict__ botkey, float* __restrict__ botval) {
  __shared__ float sTK[64 * 32]; __shared__ int sTI[64 * 32];
  __shared__ float sBK[64 * 32]; __shared__ int sBI[64 * 32];
  const int t = threadIdx.x;
  const int h = blockIdx.x;
  float* myTK = sTK + t * 32; int* myTI = sTI + t * 32;
  float* myBK = sBK + t * 32; int* myBI = sBI + t * 32;
  for (int i = 0; i < 32; ++i) {
    myTK[i] = -1e30f; myTI[i] = 0;
    myBK[i] =  1e30f; myBI[i] = 0;
  }
  for (int n = t; n < NTOK; n += 64) {
    const float* kp = qkv + (size_t)n * 384 + 128 + h;
    if (n + 64 < NTOK) __builtin_prefetch(kp + (size_t)64 * 384, 0, 1);
    const float k = *kp;
    if (k > myTK[31]) {              // descending insert
      int p = 31;
      while (p > 0 && myTK[p - 1] < k) { myTK[p] = myTK[p - 1]; myTI[p] = myTI[p - 1]; --p; }
      myTK[p] = k; myTI[p] = n;
    }
    if (k < myBK[31]) {              // ascending insert
      int p = 31;
      while (p > 0 && myBK[p - 1] > k) { myBK[p] = myBK[p - 1]; myBI[p] = myBI[p - 1]; --p; }
      myBK[p] = k; myBI[p] = n;
    }
  }
  __syncthreads();
  if (t == 0) {                       // merge top
    int pos[64]; for (int c = 0; c < 64; ++c) pos[c] = 0;
    for (int r = 0; r < 32; ++r) {
      float best = -1e30f; int bc = 0;
      for (int c = 0; c < 64; ++c) {
        float cand = sTK[c * 32 + pos[c]];
        if (cand > best) { best = cand; bc = c; }
      }
      const int idx = sTI[bc * 32 + pos[bc]]; pos[bc]++;
      topkey[h * 32 + r] = best;
      topval[h * 32 + r] = qkv[(size_t)idx * 384 + 256 + h];
    }
  }
  if (t == 1) {                       // merge bottom
    int pos[64]; for (int c = 0; c < 64; ++c) pos[c] = 0;
    for (int r = 0; r < 32; ++r) {
      float best = 1e30f; int bc = 0;
      for (int c = 0; c < 64; ++c) {
        float cand = sBK[c * 32 + pos[c]];
        if (cand < best) { best = cand; bc = c; }
      }
      const int idx = sBI[bc * 32 + pos[bc]]; pos[bc]++;
      botkey[h * 32 + r] = best;
      botval[h * 32 + r] = qkv[(size_t)idx * 384 + 256 + h];
    }
  }
}

// ---------------------------------------------------------------------------
// Attention apply: per (n,h): scalar q, pick top or bot table by sign(q),
// 32-wide stable softmax, weighted sum of scalar values. Tables for 64 heads
// cached in LDS (32KB); blockIdx.y selects head half; grid-stride over n.
// Output bf16 (A operand of the Wo GEMM).
// ---------------------------------------------------------------------------
__global__ void attn_kernel(const float* __restrict__ qkv,
                            const float* __restrict__ topkey, const float* __restrict__ topval,
                            const float* __restrict__ botkey, const float* __restrict__ botval,
                            bf16_t* __restrict__ attn) {
  __shared__ float sTK[64 * 32], sTV[64 * 32], sBK[64 * 32], sBV[64 * 32];
  const int hy = blockIdx.y * 64;
  for (int i = threadIdx.x; i < 64 * 32; i += 256) {
    sTK[i] = topkey[hy * 32 + i]; sTV[i] = topval[hy * 32 + i];
    sBK[i] = botkey[hy * 32 + i]; sBV[i] = botval[hy * 32 + i];
  }
  __syncthreads();
  const int lh = threadIdx.x & 63;
  const int h  = hy + lh;
  for (size_t n = (size_t)blockIdx.x * 4 + (threadIdx.x >> 6); n < NTOK;
       n += (size_t)gridDim.x * 4) {
    const float q = qkv[n * 384 + h];
    const float* kk = (q > 0.f) ? (sTK + lh * 32) : (sBK + lh * 32);
    const float* vv = (q > 0.f) ? (sTV + lh * 32) : (sBV + lh * 32);
    float m = -1e30f;
#pragma unroll
    for (int k = 0; k < TOPK; ++k) { float s = q * kk[k]; m = fmaxf(m, s); }
    float den = 0.f, num = 0.f;
#pragma unroll
    for (int k = 0; k < TOPK; ++k) {
      float p = __expf(q * kk[k] - m);
      den += p; num += p * vv[k];
    }
    attn[n * DIMF + h] = (bf16_t)(num / den);
  }
}

// ---------------------------------------------------------------------------
// Weight convert/pad f32 -> bf16.  dst is dRows x dK; src is sRows x sK;
// out-of-range entries zero-filled (HIDDEN 341 -> 352 padding).
// ---------------------------------------------------------------------------
__global__ void convpad_kernel(const float* __restrict__ src, bf16_t* __restrict__ dst,
                               int dRows, int dK, int sRows, int sK) {
  const size_t tid = (size_t)blockIdx.x * blockDim.x + threadIdx.x;
  if (tid >= (size_t)dRows * dK) return;
  const int r = (int)(tid / dK), k = (int)(tid % dK);
  dst[tid] = (r < sRows && k < sK) ? (bf16_t)src[(size_t)r * sK + k] : (bf16_t)0.f;
}

// ---------------------------------------------------------------------------
extern "C" void kernel_launch(void* const* d_in, const int* in_sizes, int n_in,
                              void* d_out, int out_size, void* d_ws, size_t ws_size,
                              hipStream_t stream) {
  (void)in_sizes; (void)n_in; (void)out_size; (void)ws_size;
  const int*   bytes = (const int*)  d_in[0];
  const float* be    = (const float*)d_in[1];
  const float* bpe   = (const float*)d_in[2];
  const float* lpe   = (const float*)d_in[3];
  const float* Wq    = (const float*)d_in[4];
  const float* Wk    = (const float*)d_in[5];
  const float* Wv    = (const float*)d_in[6];
  const float* Wo    = (const float*)d_in[7];
  const float* n1w   = (const float*)d_in[8];
  const float* n2w   = (const float*)d_in[9];
  const float* wg    = (const float*)d_in[10];
  const float* wu    = (const float*)d_in[11];
  const float* wd    = (const float*)d_in[12];
  const float* fnw   = (const float*)d_in[13];
  const float* outw  = (const float*)d_in[14];
  float* out = (float*)d_out;

  char* ws = (char*)d_ws;
  size_t off = 0;
  auto take = [&](size_t bytes_) { char* p = ws + off; off = (off + bytes_ + 255) & ~(size_t)255; return p; };
  float*  x     = (float*) take((size_t)NTOK * 128 * 4);   // residual stream
  bf16_t* xn    = (bf16_t*)take((size_t)NTOK * 128 * 2);   // normalized activations
  float*  qkv   = (float*) take((size_t)NTOK * 384 * 4);   // q|k|v per head
  bf16_t* attn  = (bf16_t*)take((size_t)NTOK * 128 * 2);   // pre-Wo attention out
  bf16_t* hbuf  = (bf16_t*)take((size_t)NTOK * HIDPAD * 2);// swiglu hidden
  bf16_t* wqkv  = (bf16_t*)take((size_t)2 * 384 * 128 * 2);
  bf16_t* wo    = (bf16_t*)take((size_t)2 * 128 * 128 * 2);
  bf16_t* wgu   = (bf16_t*)take((size_t)2 * 2 * HIDPAD * 128 * 2);
  bf16_t* wdn   = (bf16_t*)take((size_t)2 * 128 * HIDPAD * 2);
  bf16_t* wout  = (bf16_t*)take((size_t)256 * 128 * 2);
  float*  tkK   = (float*) take((size_t)128 * 32 * 4);
  float*  tkV   = (float*) take((size_t)128 * 32 * 4);
  float*  tbK   = (float*) take((size_t)128 * 32 * 4);
  float*  tbV   = (float*) take((size_t)128 * 32 * 4);

  auto conv = [&](const float* s, bf16_t* d, int dR, int dK, int sR, int sK) {
    int tot = dR * dK;
    convpad_kernel<<<(tot + 255) / 256, 256, 0, stream>>>(s, d, dR, dK, sR, sK);
  };

  // ---- weight prep (bf16, padded) ----
  for (int l = 0; l < 2; ++l) {
    bf16_t* wqkv_l = wqkv + (size_t)l * 384 * 128;
    conv(Wq + (size_t)l * 128 * 128, wqkv_l,             128, 128, 128, 128);
    conv(Wk + (size_t)l * 128 * 128, wqkv_l + 128 * 128, 128, 128, 128, 128);
    conv(Wv + (size_t)l * 128 * 128, wqkv_l + 256 * 128, 128, 128, 128, 128);
    conv(Wo + (size_t)l * 128 * 128, wo + (size_t)l * 128 * 128, 128, 128, 128, 128);
    bf16_t* wgu_l = wgu + (size_t)l * 2 * HIDPAD * 128;
    conv(wg + (size_t)l * HIDRAW * 128, wgu_l,                HIDPAD, 128, HIDRAW, 128);
    conv(wu + (size_t)l * HIDRAW * 128, wgu_l + HIDPAD * 128, HIDPAD, 128, HIDRAW, 128);
    conv(wd + (size_t)l * 128 * HIDRAW, wdn + (size_t)l * 128 * HIDPAD, 128, HIDPAD, 128, HIDRAW);
  }
  conv(outw, wout, 256, 128, 256, 128);

  // ---- embedding ----
  embed_kernel<<<(NTOK * 32) / 256, 256, 0, stream>>>(bytes, be, bpe, lpe, x);

  const int gemmGrid = NTOK / 128;      // 2048 blocks x 8 waves x 16 rows
  const int normGrid = NTOK / 8;

  for (int l = 0; l < 2; ++l) {
    // attention sub-block
    rmsnorm_kernel<<<normGrid, 256, 0, stream>>>(x, n1w + (size_t)l * 128, xn);
    gemm_kernel<128, 384, 0><<<gemmGrid, 256, 0, stream>>>(
        xn, wqkv + (size_t)l * 384 * 128, qkv, nullptr, 384);
    topk_kernel<<<HEADS, 64, 0, stream>>>(qkv, tkK, tkV, tbK, tbV);
    attn_kernel<<<dim3(2048, 2), 256, 0, stream>>>(qkv, tkK, tkV, tbK, tbV, attn);
    gemm_kernel<128, 128, 1><<<gemmGrid, 256, 0, stream>>>(
        attn, wo + (size_t)l * 128 * 128, x, nullptr, 128);   // x += attn @ Wo^T

    // swiglu sub-block
    rmsnorm_kernel<<<normGrid, 256, 0, stream>>>(x, n2w + (size_t)l * 128, xn);
    gemm_kernel<128, HIDPAD, 2><<<gemmGrid, 256, 0, stream>>>(
        xn, wgu + (size_t)l * 2 * HIDPAD * 128, nullptr, hbuf, HIDPAD);
    gemm_kernel<HIDPAD, 128, 1><<<gemmGrid, 256, 0, stream>>>(
        hbuf, wdn + (size_t)l * 128 * HIDPAD, x, nullptr, 128); // x += h @ Wd^T
  }

  // ---- final norm + output projection ----
  rmsnorm_kernel<<<normGrid, 256, 0, stream>>>(x, fnw, xn);
  gemm_kernel<128, 256, 0><<<gemmGrid, 256, 0, stream>>>(xn, wout, out, nullptr, 256);
}
